// GRUCell_78451872628868
// MI455X (gfx1250) — matrix operand order repaired and checked by reading, not compiled, not run on hardware
//
#include <hip/hip_runtime.h>

#define BATCH  8192
#define HID    1024
#define CONCAT 2048

typedef __attribute__((ext_vector_type(16))) __bf16 bf16x16;
typedef __attribute__((ext_vector_type(8)))  __bf16 bf16x8;
typedef __attribute__((ext_vector_type(8)))  float  f32x8;

static __device__ __forceinline__ unsigned short f32_to_bf16_rn(float f) {
    unsigned int u = __float_as_uint(f);
    unsigned int r = u + 0x7FFFu + ((u >> 16) & 1u);
    return (unsigned short)(r >> 16);
}

// ---------------- conversion kernels ----------------

// concat1[row][0:1024) = bf16(h_prev), concat1[row][1024:2048) = bf16(x)
// concat2[row][1024:2048) = bf16(x)   (first half written later by gate kernel)
__global__ __launch_bounds__(256)
void build_concat_kernel(const float* __restrict__ x,
                         const float* __restrict__ h_prev,
                         unsigned short* __restrict__ concat1,
                         unsigned short* __restrict__ concat2) {
    int e   = (blockIdx.x * blockDim.x + threadIdx.x) * 4;   // 4 elements / thread
    int row = e >> 11;            // / 2048
    int col = e & (CONCAT - 1);
    const float* src = (col < HID) ? (h_prev + (size_t)row * HID + col)
                                   : (x      + (size_t)row * HID + (col - HID));
    float4 v = *(const float4*)src;
    ushort4 o;
    o.x = f32_to_bf16_rn(v.x);
    o.y = f32_to_bf16_rn(v.y);
    o.z = f32_to_bf16_rn(v.z);
    o.w = f32_to_bf16_rn(v.w);
    *(ushort4*)(concat1 + (size_t)e) = o;
    if (col >= HID) *(ushort4*)(concat2 + (size_t)e) = o;
}

__global__ __launch_bounds__(256)
void convert_weights_kernel(const float* __restrict__ wz,
                            const float* __restrict__ wr,
                            const float* __restrict__ wh,
                            unsigned short* __restrict__ wzb,
                            unsigned short* __restrict__ wrb,
                            unsigned short* __restrict__ whb) {
    size_t e = (size_t)(blockIdx.x * blockDim.x + threadIdx.x) * 4;
    float4 a = *(const float4*)(wz + e);
    float4 b = *(const float4*)(wr + e);
    float4 c = *(const float4*)(wh + e);
    ushort4 oa, ob, oc;
    oa.x = f32_to_bf16_rn(a.x); oa.y = f32_to_bf16_rn(a.y);
    oa.z = f32_to_bf16_rn(a.z); oa.w = f32_to_bf16_rn(a.w);
    ob.x = f32_to_bf16_rn(b.x); ob.y = f32_to_bf16_rn(b.y);
    ob.z = f32_to_bf16_rn(b.z); ob.w = f32_to_bf16_rn(b.w);
    oc.x = f32_to_bf16_rn(c.x); oc.y = f32_to_bf16_rn(c.y);
    oc.z = f32_to_bf16_rn(c.z); oc.w = f32_to_bf16_rn(c.w);
    *(ushort4*)(wzb + e) = oa;
    *(ushort4*)(wrb + e) = ob;
    *(ushort4*)(whb + e) = oc;
}

// ---------------- WMMA fragment helpers ----------------
// A 16x32 bf16 (ISA 7.12.2): lane(half=0) K = [k+0..7] and [k+16..23];
//                            lane(half=1) K = [k+8..15] and [k+24..31]
static __device__ __forceinline__ bf16x16
load_frag_a(const unsigned short* __restrict__ base, int row, int k, int ld, int hf) {
    const __bf16* p = (const __bf16*)(base + (size_t)row * ld + k + hf * 8);
    bf16x8 lo = *(const bf16x8*)p;          // K = k + hf*8 .. +7
    bf16x8 hi = *(const bf16x8*)(p + 16);   // K = k + 16 + hf*8 .. +7
    return __builtin_shufflevector(lo, hi, 0,1,2,3,4,5,6,7,8,9,10,11,12,13,14,15);
}

// B 32x16 bf16: lanes 0-15 hold K=[k..k+7],[k+16..k+23]; lanes 16-31 hold K=[k+8..k+15],[k+24..k+31],
// col = lane%16. B[k][n] = W[n][k] (row-major W) -> K-contiguous read of W row n.
static __device__ __forceinline__ bf16x16
load_frag_b(const unsigned short* __restrict__ base, int col, int k, int ld, int hf) {
    const __bf16* p = (const __bf16*)(base + (size_t)col * ld + k + hf * 8);
    bf16x8 lo = *(const bf16x8*)p;
    bf16x8 hi = *(const bf16x8*)(p + 16);
    return __builtin_shufflevector(lo, hi, 0,1,2,3,4,5,6,7,8,9,10,11,12,13,14,15);
}

// ---------------- gate kernel: z and r (shared A) ----------------
// Block tile 128(M) x 128(N), 8 waves as 2(M) x 4(N); wave tile 64x32 per gate.
// Per k-step / wave: 8 KB loaded, 16 WMMAs -> 32 FLOP/byte through VMEM.
__global__ __launch_bounds__(256)
void gru_gates_kernel(const unsigned short* __restrict__ concat1,
                      const unsigned short* __restrict__ wz,
                      const unsigned short* __restrict__ wr,
                      const float* __restrict__ b_z,
                      const float* __restrict__ b_r,
                      const float* __restrict__ h_prev,
                      float* __restrict__ zbuf,
                      unsigned short* __restrict__ concat2) {
    const int lane = threadIdx.x & 31;
    const int wave = threadIdx.x >> 5;
    const int hf   = lane >> 4;
    const int ln   = lane & 15;
    const int wm   = wave & 1;             // 2 waves along M
    const int wn   = wave >> 1;            // 4 waves along N
    const int rowBase = blockIdx.y * 128 + wm * 64;   // 64 rows / wave
    const int colBase = blockIdx.x * 128 + wn * 32;   // 32 cols / wave

    f32x8 accZ[4][2], accR[4][2];
#pragma unroll
    for (int m = 0; m < 4; ++m)
#pragma unroll
        for (int n = 0; n < 2; ++n) { accZ[m][n] = (f32x8)(0.0f); accR[m][n] = (f32x8)(0.0f); }

    for (int k = 0; k < CONCAT; k += 32) {
        if (k + 32 < CONCAT) {   // L2-resident working set: keep the pipes fed
            __builtin_prefetch(concat1 + (size_t)(rowBase + ln) * CONCAT + k + 32, 0, 3);
            __builtin_prefetch(wz + (size_t)(colBase + ln) * CONCAT + k + 32, 0, 3);
            __builtin_prefetch(wr + (size_t)(colBase + ln) * CONCAT + k + 32, 0, 3);
        }
        bf16x16 a[4];
#pragma unroll
        for (int m = 0; m < 4; ++m)
            a[m] = load_frag_a(concat1, rowBase + m * 16 + ln, k, CONCAT, hf);
#pragma unroll
        for (int n = 0; n < 2; ++n) {
            bf16x16 bzf = load_frag_b(wz, colBase + n * 16 + ln, k, CONCAT, hf);
            bf16x16 brf = load_frag_b(wr, colBase + n * 16 + ln, k, CONCAT, hf);
#pragma unroll
            for (int m = 0; m < 4; ++m) {
                accZ[m][n] = __builtin_amdgcn_wmma_f32_16x16x32_bf16(
                    false, a[m], false, bzf, (short)0, accZ[m][n], false, false);
                accR[m][n] = __builtin_amdgcn_wmma_f32_16x16x32_bf16(
                    false, a[m], false, brf, (short)0, accR[m][n], false, false);
            }
        }
    }

    // epilogue: sigmoid, write z (f32) and bf16(r*h_prev) into concat2[:, 0:H)
#pragma unroll
    for (int n = 0; n < 2; ++n) {
        const int col = colBase + n * 16 + ln;
        const float bz = b_z[col];
        const float br = b_r[col];
#pragma unroll
        for (int m = 0; m < 4; ++m) {
#pragma unroll
            for (int v = 0; v < 8; ++v) {
                const int row = rowBase + m * 16 + hf * 8 + v;
                const float zv = 1.0f / (1.0f + __expf(-(accZ[m][n][v] + bz)));
                const float rv = 1.0f / (1.0f + __expf(-(accR[m][n][v] + br)));
                zbuf[(size_t)row * HID + col] = zv;
                const float rh = rv * h_prev[(size_t)row * HID + col];
                concat2[(size_t)row * CONCAT + col] = f32_to_bf16_rn(rh);
            }
        }
    }
}

// ---------------- candidate kernel: h_tilde + final blend ----------------
// Block tile 256(M) x 128(N), 8 waves as 4(M) x 2(N); wave tile 64x64.
// Per k-step / wave: 8 KB loaded, 16 WMMAs -> 32 FLOP/byte (matches gate kernel).
__global__ __launch_bounds__(256)
void gru_candidate_kernel(const unsigned short* __restrict__ concat2,
                          const unsigned short* __restrict__ wh,
                          const float* __restrict__ b_h,
                          const float* __restrict__ zbuf,
                          const float* __restrict__ h_prev,
                          float* __restrict__ out) {
    const int lane = threadIdx.x & 31;
    const int wave = threadIdx.x >> 5;
    const int hf   = lane >> 4;
    const int ln   = lane & 15;
    const int wm   = wave & 3;             // 4 waves along M
    const int wn   = wave >> 2;            // 2 waves along N
    const int rowBase = blockIdx.y * 256 + wm * 64;   // 64 rows / wave
    const int colBase = blockIdx.x * 128 + wn * 64;   // 64 cols / wave

    f32x8 acc[4][4];
#pragma unroll
    for (int m = 0; m < 4; ++m)
#pragma unroll
        for (int n = 0; n < 4; ++n) acc[m][n] = (f32x8)(0.0f);

    for (int k = 0; k < CONCAT; k += 32) {
        if (k + 32 < CONCAT) {
            __builtin_prefetch(concat2 + (size_t)(rowBase + ln) * CONCAT + k + 32, 0, 3);
            __builtin_prefetch(wh + (size_t)(colBase + ln) * CONCAT + k + 32, 0, 3);
        }
        bf16x16 a[4];
#pragma unroll
        for (int m = 0; m < 4; ++m)
            a[m] = load_frag_a(concat2, rowBase + m * 16 + ln, k, CONCAT, hf);
#pragma unroll
        for (int n = 0; n < 4; ++n) {
            bf16x16 bf = load_frag_b(wh, colBase + n * 16 + ln, k, CONCAT, hf);
#pragma unroll
            for (int m = 0; m < 4; ++m)
                acc[m][n] = __builtin_amdgcn_wmma_f32_16x16x32_bf16(
                    false, a[m], false, bf, (short)0, acc[m][n], false, false);
        }
    }

#pragma unroll
    for (int n = 0; n < 4; ++n) {
        const int col = colBase + n * 16 + ln;
        const float bh = b_h[col];
#pragma unroll
        for (int m = 0; m < 4; ++m) {
#pragma unroll
            for (int v = 0; v < 8; ++v) {
                const int row = rowBase + m * 16 + hf * 8 + v;
                const float ht = tanhf(acc[m][n][v] + bh);
                const size_t idx = (size_t)row * HID + col;
                const float z = zbuf[idx];
                const float hp = h_prev[idx];
                out[idx] = (1.0f - z) * hp + z * ht;
            }
        }
    }
}

// ---------------- launch ----------------
extern "C" void kernel_launch(void* const* d_in, const int* in_sizes, int n_in,
                              void* d_out, int out_size, void* d_ws, size_t ws_size,
                              hipStream_t stream) {
    const float* x      = (const float*)d_in[0];
    const float* h_prev = (const float*)d_in[1];
    const float* W_z    = (const float*)d_in[2];
    const float* b_z    = (const float*)d_in[3];
    const float* W_r    = (const float*)d_in[4];
    const float* b_r    = (const float*)d_in[5];
    const float* W_h    = (const float*)d_in[6];
    const float* b_h    = (const float*)d_in[7];
    float* out = (float*)d_out;

    char* ws = (char*)d_ws;
    unsigned short* concat1 = (unsigned short*)ws; ws += (size_t)BATCH * CONCAT * 2;
    unsigned short* concat2 = (unsigned short*)ws; ws += (size_t)BATCH * CONCAT * 2;
    unsigned short* wzb     = (unsigned short*)ws; ws += (size_t)HID * CONCAT * 2;
    unsigned short* wrb     = (unsigned short*)ws; ws += (size_t)HID * CONCAT * 2;
    unsigned short* whb     = (unsigned short*)ws; ws += (size_t)HID * CONCAT * 2;
    float*          zbuf    = (float*)ws;          ws += (size_t)BATCH * HID * 4;

    // 1) f32 -> bf16 staging
    build_concat_kernel<<<(BATCH * CONCAT / 4) / 256, 256, 0, stream>>>(
        x, h_prev, concat1, concat2);
    convert_weights_kernel<<<(HID * CONCAT / 4) / 256, 256, 0, stream>>>(
        W_z, W_r, W_h, wzb, wrb, whb);

    // 2) fused z/r dual-GEMM
    dim3 gridG(HID / 128, BATCH / 128);   // 8 x 64 blocks, 256 threads (8 waves)
    gru_gates_kernel<<<gridG, 256, 0, stream>>>(
        concat1, wzb, wrb, b_z, b_r, h_prev, zbuf, concat2);

    // 3) candidate GEMM + final blend
    dim3 gridC(HID / 128, BATCH / 256);   // 8 x 32 blocks, 256 threads (8 waves)
    gru_candidate_kernel<<<gridC, 256, 0, stream>>>(
        concat2, whb, b_h, zbuf, h_prev, out);
}
